// StandardKernelModel_89412629168319
// MI455X (gfx1250) — matrix operand order repaired and checked by reading, not compiled, not run on hardware
//
#include <hip/hip_runtime.h>

// ---------------------------------------------------------------------------
// Laplacian-kernel regression, fused two-stage bf16 WMMA pipeline (gfx1250)
// with Tensor-Data-Mover (TDM) staging of all WMMA operands through LDS.
//   pred = exp(-||x - z|| / bw) @ W
// N=8192, M=20000 (padded 20096), D=1024, Y=256.
// ---------------------------------------------------------------------------

#define D_DIM   1024
#define N_ROWS  8192
#define M_REAL  20000
#define M_TILES 157
#define M_PAD   (M_TILES * 128)   // 20096
#define Y_DIM   256
#define SPLIT_M 8
#define TILES_PER_CHUNK ((M_TILES + SPLIT_M - 1) / SPLIT_M)  // 20
#define KC      128               // K-chunk staged per TDM transfer
#define NCHUNK  (D_DIM / KC)      // 8
#define LDK     136               // LDS row stride in bf16 elements (TDM pad)

typedef __attribute__((ext_vector_type(16))) __bf16        v16bf;
typedef __attribute__((ext_vector_type(8)))  float         v8f;
typedef __attribute__((ext_vector_type(4)))  unsigned int  v4u;
typedef __attribute__((ext_vector_type(8)))  int           v8i;
typedef __attribute__((ext_vector_type(4)))  int           v4i;

union Frag {
    v16bf v;
    v4u   q[2];
};

__device__ __forceinline__ unsigned short f2bf(float f) {
    unsigned int u = __float_as_uint(f);
    u += 0x7fffu + ((u >> 16) & 1u);   // round-to-nearest-even
    return (unsigned short)(u >> 16);
}

#if __has_builtin(__builtin_amdgcn_tensor_load_to_lds) && \
    __has_builtin(__builtin_amdgcn_s_wait_tensorcnt)
#define HAVE_TDM 1
#else
#define HAVE_TDM 0
#endif

#if HAVE_TDM
// Issue one 2D TDM tile load: `height` rows of `width` bf16 elements, source
// row stride `src_stride` elements, packed into LDS with +4-DWORD padding
// every 64 DWORDs (row stride becomes width+8 = 136 elements for width=128).
__device__ __forceinline__ void tdm_load_2d(
    unsigned int lds_addr, const void* gsrc,
    unsigned int width, unsigned int height, unsigned int src_stride)
{
    unsigned long long ga = (unsigned long long)gsrc;
    v4u g0;
    g0[0] = 1u;                                    // count=1, user descriptor
    g0[1] = lds_addr;                              // LDS byte address
    g0[2] = (unsigned int)ga;                      // global_addr[31:0]
    g0[3] = (unsigned int)((ga >> 32) & 0x01ffffffu) | 0x80000000u; // [56:32], type=2
    v8i g1;
    // data_size=1 (2B), pad_enable=1, pad_interval=5 (64 DW), pad_amount=3 (4 DW)
    g1[0] = (int)((1u << 16) | (1u << 20) | (5u << 22) | (3u << 25));
    g1[1] = (int)((width & 0xffffu) << 16);                             // tensor_dim0 lo
    g1[2] = (int)(((width >> 16) & 0xffffu) | ((height & 0xffffu) << 16)); // dim0 hi | dim1 lo
    g1[3] = (int)(((height >> 16) & 0xffffu) | ((width & 0xffffu) << 16)); // dim1 hi | tile_dim0
    g1[4] = (int)(height & 0xffffu);                                    // tile_dim1 (tile_dim2=0)
    g1[5] = (int)src_stride;                                            // tensor_dim0_stride lo32
    g1[6] = 0;                                                          // stride0 hi | stride1 lo
    g1[7] = 0;
    v4i z4 = {0, 0, 0, 0};
#if defined(__clang_major__) && (__clang_major__ >= 23)
    v8i z8 = {0, 0, 0, 0, 0, 0, 0, 0};
    __builtin_amdgcn_tensor_load_to_lds(g0, g1, z4, z4, z8, 0);
#else
    __builtin_amdgcn_tensor_load_to_lds(g0, g1, z4, z4, 0);
#endif
}
#endif

// Fallback / utility: cooperative copy of a 2D tile into LDS with stride LDK.
__device__ __forceinline__ void coop_copy_tile(
    unsigned short* __restrict__ dst, const unsigned short* __restrict__ src,
    int rows, int src_stride, int nthreads)
{
    const int chunks_per_row = KC / 8;             // b128 chunks
    const int total = rows * chunks_per_row;
    for (int i = threadIdx.x; i < total; i += nthreads) {
        const int r = i / chunks_per_row;
        const int c = (i % chunks_per_row) * 8;
        *(v4u*)(dst + r * LDK + c) = *(const v4u*)(src + (size_t)r * src_stride + c);
    }
}

// ---- prep: fp32 rows -> bf16 rows + fp32 squared norms (zero-pads) --------
__global__ __launch_bounds__(256) void prep_rows_kernel(
    const float* __restrict__ src, unsigned short* __restrict__ dst,
    float* __restrict__ sq, int valid_rows)
{
    const int row = blockIdx.x;
    const int t   = threadIdx.x;
    float acc = 0.0f;
    unsigned short* d = dst + (size_t)row * D_DIM;
    if (row < valid_rows) {
        const float* s = src + (size_t)row * D_DIM;
        for (int j = t; j < D_DIM; j += 256) {
            float v = s[j];
            d[j] = f2bf(v);
            acc += v * v;
        }
    } else {
        for (int j = t; j < D_DIM; j += 256) d[j] = 0;
    }
    __shared__ float red[256];
    red[t] = acc;
    __syncthreads();
    for (int s2 = 128; s2 > 0; s2 >>= 1) {
        if (t < s2) red[t] += red[t + s2];
        __syncthreads();
    }
    if (t == 0) sq[row] = red[0];
}

// ---- prep: weight [M][Y] fp32 -> weightT [Y][M_PAD] bf16 (zero-padded) ----
__global__ __launch_bounds__(256) void prep_weightT_kernel(
    const float* __restrict__ w, unsigned short* __restrict__ wtT)
{
    __shared__ unsigned short tile[32][33];
    const int mb = blockIdx.x * 32;
    const int yb = blockIdx.y * 32;
    const int tx = threadIdx.x;   // 0..31
    const int ty = threadIdx.y;   // 0..7
    for (int i = ty; i < 32; i += 8) {
        const int m = mb + i;
        float v = (m < M_REAL) ? w[(size_t)m * Y_DIM + (yb + tx)] : 0.0f;
        tile[i][tx] = f2bf(v);
    }
    __syncthreads();
    for (int i = ty; i < 32; i += 8) {
        wtT[(size_t)(yb + i) * M_PAD + (mb + tx)] = tile[tx][i];
    }
}

// ---- fused main kernel ----------------------------------------------------
// grid = (N/128, SPLIT_M), block = 512 threads = 16 waves (wave32).
// GEMM1: 4x4 wave grid, wave (wr,wc) computes 32x32 of the 128x128 kmat tile.
// GEMM2: wave w: rows 16*(w&7).., Y columns 128*(w>>3)..
// All WMMA operands staged in LDS (TDM double-buffered K chunks + W tile).
__global__ __launch_bounds__(512, 1) void laplacian_fused_kernel(
    const unsigned short* __restrict__ Xb,   // [N_ROWS][D] bf16
    const unsigned short* __restrict__ Zb,   // [M_PAD][D] bf16
    const unsigned short* __restrict__ WtT,  // [Y_DIM][M_PAD] bf16
    const float* __restrict__ xsq,           // [N_ROWS]
    const float* __restrict__ zsq,           // [M_PAD]
    float* __restrict__ partial,             // [SPLIT_M][N_ROWS*Y_DIM]
    const int* __restrict__ bw_ptr)
{
    extern __shared__ unsigned short smem[];
    unsigned short* sX[2];                    // [128][LDK] bf16 each
    unsigned short* sZ[2];
    sX[0] = smem;
    sX[1] = smem + 128 * LDK;
    sZ[0] = smem + 2 * 128 * LDK;
    sZ[1] = smem + 3 * 128 * LDK;
    unsigned short* sW = smem + 4 * 128 * LDK;             // [256][LDK]
    unsigned short* sK = sW + 256 * LDK;                   // [128][LDK]

    const int ntile  = blockIdx.x;
    const int schunk = blockIdx.y;
    const int tile_lo = schunk * TILES_PER_CHUNK;
    int tile_hi = tile_lo + TILES_PER_CHUNK;
    if (tile_hi > M_TILES) tile_hi = M_TILES;

    const int lane = threadIdx.x & 31;
    const int w    = threadIdx.x >> 5;    // 0..15
    const int wr   = w >> 2;              // GEMM1 row group (32 rows)
    const int wc   = w & 3;               // GEMM1 col group (32 cols)
    const int r1   = w & 7;               // GEMM2 16-row group
    const int cg   = w >> 3;              // GEMM2 Y half

    const int ln = lane & 15;
    const int kh = lane >> 4;

    const float inv_bw = 1.0f / (float)bw_ptr[0];
    const int xrowbase = ntile * 128;

    // pred accumulators: 16 N-rows x 128 Y-cols = 8 C tiles
    v8f c2[8];
    #pragma unroll
    for (int i = 0; i < 8; ++i)
        #pragma unroll
        for (int e = 0; e < 8; ++e) c2[i][e] = 0.0f;

    for (int mt = tile_lo; mt < tile_hi; ++mt) {
        const int zbase = mt * 128;

#if HAVE_TDM
        if (w == 0) {   // one wave drives the TDM (EXEC ignored by tensor ops)
            tdm_load_2d((unsigned int)(size_t)sW, WtT + zbase, KC, 256, M_PAD);
            tdm_load_2d((unsigned int)(size_t)sX[0],
                        Xb + (size_t)xrowbase * D_DIM, KC, 128, D_DIM);
            tdm_load_2d((unsigned int)(size_t)sZ[0],
                        Zb + (size_t)zbase * D_DIM, KC, 128, D_DIM);
        }
#endif
        // ---------------- GEMM1: c1 = X . Z^T over K = D ------------------
        v8f c1[2][2];
        #pragma unroll
        for (int i = 0; i < 2; ++i)
            #pragma unroll
            for (int j = 0; j < 2; ++j)
                #pragma unroll
                for (int e = 0; e < 8; ++e) c1[i][j][e] = 0.0f;

        for (int c = 0; c < NCHUNK; ++c) {
            const int cur = c & 1;
#if HAVE_TDM
            if (w == 0) {
                if (c + 1 < NCHUNK) {
                    const int nxt = cur ^ 1;
                    tdm_load_2d((unsigned int)(size_t)sX[nxt],
                                Xb + (size_t)xrowbase * D_DIM + (c + 1) * KC,
                                KC, 128, D_DIM);
                    tdm_load_2d((unsigned int)(size_t)sZ[nxt],
                                Zb + (size_t)zbase * D_DIM + (c + 1) * KC,
                                KC, 128, D_DIM);
                    __builtin_amdgcn_s_wait_tensorcnt(2);  // chunk c (+W) landed
                } else {
                    __builtin_amdgcn_s_wait_tensorcnt(0);
                }
            }
            __syncthreads();                 // staged data visible to all waves
#else
            __syncthreads();
            if (c == 0 && mt == tile_lo)
                coop_copy_tile(sW, WtT + zbase, 256, M_PAD, 512);
            else if (c == 0)
                coop_copy_tile(sW, WtT + zbase, 256, M_PAD, 512);
            coop_copy_tile(sX[cur], Xb + (size_t)xrowbase * D_DIM + c * KC, 128, D_DIM, 512);
            coop_copy_tile(sZ[cur], Zb + (size_t)zbase * D_DIM + c * KC, 128, D_DIM, 512);
            __syncthreads();
#endif
            const unsigned short* xb = sX[cur];
            const unsigned short* zb = sZ[cur];
            #pragma unroll
            for (int ko = 0; ko < KC; ko += 32) {
                Frag a[2], b[2];
                #pragma unroll
                for (int i = 0; i < 2; ++i) {
                    const unsigned short* ap =
                        xb + (32 * wr + 16 * i + ln) * LDK + ko + 8 * kh;
                    a[i].q[0] = *(const v4u*)ap;
                    a[i].q[1] = *(const v4u*)(ap + 16);
                }
                #pragma unroll
                for (int j = 0; j < 2; ++j) {
                    const unsigned short* bp =
                        zb + (32 * wc + 16 * j + ln) * LDK + ko + 8 * kh;
                    b[j].q[0] = *(const v4u*)bp;
                    b[j].q[1] = *(const v4u*)(bp + 16);
                }
                #pragma unroll
                for (int i = 0; i < 2; ++i)
                    #pragma unroll
                    for (int j = 0; j < 2; ++j)
                        c1[i][j] = __builtin_amdgcn_wmma_f32_16x16x32_bf16(
                            false, a[i].v, false, b[j].v, (short)0, c1[i][j],
                            false, false);
            }
            __syncthreads();                 // done with buffer before refill
        }

        // -------- epilogue: d2 -> exp(-d/bw), bf16 kmat tile into LDS -----
        // C layout: element (VGPR r, lane) -> row = r + 8*kh, col = ln.
        float xsl[2][8];
        #pragma unroll
        for (int i = 0; i < 2; ++i)
            #pragma unroll
            for (int r = 0; r < 8; ++r)
                xsl[i][r] = xsq[xrowbase + 32 * wr + 16 * i + 8 * kh + r];
        float zsl[2];
        #pragma unroll
        for (int j = 0; j < 2; ++j)
            zsl[j] = zsq[zbase + 32 * wc + 16 * j + ln];

        #pragma unroll
        for (int i = 0; i < 2; ++i)
            #pragma unroll
            for (int j = 0; j < 2; ++j) {
                const int lcol = 32 * wc + 16 * j + ln;
                #pragma unroll
                for (int r = 0; r < 8; ++r) {
                    float d2 = xsl[i][r] + zsl[j] - 2.0f * c1[i][j][r];
                    float dd = sqrtf(fmaxf(d2, 0.0f));
                    float kv = __expf(-dd * inv_bw);
                    sK[(32 * wr + 16 * i + r + 8 * kh) * LDK + lcol] = f2bf(kv);
                }
            }
        __syncthreads();

        // -------- GEMM2: c2 += kmat_tile(LDS) . W^T-tile(LDS) -------------
        #pragma unroll
        for (int ks = 0; ks < 4; ++ks) {
            Frag a2;
            const unsigned short* lp = sK + (16 * r1 + ln) * LDK + ks * 32 + 8 * kh;
            a2.q[0] = *(const v4u*)lp;
            a2.q[1] = *(const v4u*)(lp + 16);
            #pragma unroll
            for (int t = 0; t < 8; ++t) {
                const unsigned short* wp =
                    sW + (128 * cg + 16 * t + ln) * LDK + ks * 32 + 8 * kh;
                Frag b;
                b.q[0] = *(const v4u*)wp;
                b.q[1] = *(const v4u*)(wp + 16);
                c2[t] = __builtin_amdgcn_wmma_f32_16x16x32_bf16(
                    false, a2.v, false, b.v, (short)0, c2[t], false, false);
            }
        }
        __syncthreads();                     // sW/sK reusable next m-tile
    }

    // -------- write split-M partials (deterministic reduce below) ---------
    float* pout = partial + ((size_t)schunk * N_ROWS + (size_t)ntile * 128) * Y_DIM;
    #pragma unroll
    for (int t = 0; t < 8; ++t) {
        const int col = 128 * cg + 16 * t + ln;
        #pragma unroll
        for (int r = 0; r < 8; ++r) {
            const int row = 16 * r1 + r + 8 * kh;
            pout[(size_t)row * Y_DIM + col] = c2[t][r];
        }
    }
}

// ---- deterministic fixed-order reduction over SPLIT_M partials ------------
__global__ __launch_bounds__(256) void reduce_partials_kernel(
    const float* __restrict__ partial, float* __restrict__ out, int n)
{
    const int i = blockIdx.x * blockDim.x + threadIdx.x;
    if (i < n) {
        float s = 0.0f;
        #pragma unroll
        for (int c = 0; c < SPLIT_M; ++c) s += partial[(size_t)c * n + i];
        out[i] = s;
    }
}

// ---------------------------------------------------------------------------
extern "C" void kernel_launch(void* const* d_in, const int* in_sizes, int n_in,
                              void* d_out, int out_size, void* d_ws, size_t ws_size,
                              hipStream_t stream)
{
    const float* batch   = (const float*)d_in[0];   // [8192][1024] f32
    const float* centers = (const float*)d_in[1];   // [20000][1024] f32
    const float* weight  = (const float*)d_in[2];   // [20000][256] f32
    const int*   bw      = (const int*)d_in[3];     // scalar int

    char* ws = (char*)d_ws;
    auto carve = [&](size_t bytes) -> char* {
        char* p = ws;
        ws += (bytes + 255) & ~(size_t)255;
        return p;
    };

    unsigned short* Xb      = (unsigned short*)carve((size_t)N_ROWS * D_DIM * 2);   // 16 MB
    unsigned short* Zb      = (unsigned short*)carve((size_t)M_PAD  * D_DIM * 2);   // 41 MB
    unsigned short* WtT     = (unsigned short*)carve((size_t)Y_DIM  * M_PAD * 2);   // 10 MB
    float*          xsq     = (float*)carve((size_t)N_ROWS * 4);
    float*          zsq     = (float*)carve((size_t)M_PAD * 4);
    float*          partial = (float*)carve((size_t)SPLIT_M * N_ROWS * Y_DIM * 4);  // 64 MB
    (void)in_sizes; (void)n_in; (void)out_size; (void)ws_size;

    prep_rows_kernel<<<N_ROWS, 256, 0, stream>>>(batch, Xb, xsq, N_ROWS);
    prep_rows_kernel<<<M_PAD, 256, 0, stream>>>(centers, Zb, zsq, M_REAL);
    prep_weightT_kernel<<<dim3(M_PAD / 32, Y_DIM / 32), dim3(32, 8), 0, stream>>>(weight, WtT);

    // dynamic LDS: 2x X-chunk + 2x Z-chunk (34 KB each) + W tile (68 KB) + kmat (34 KB)
    const size_t lds_bytes = (size_t)(4 * 128 * LDK + 256 * LDK + 128 * LDK) * 2;   // 243712
    laplacian_fused_kernel<<<dim3(N_ROWS / 128, SPLIT_M), 512, lds_bytes, stream>>>(
        Xb, Zb, WtT, xsq, zsq, partial, bw);

    reduce_partials_kernel<<<(N_ROWS * Y_DIM + 255) / 256, 256, 0, stream>>>(
        partial, (float*)d_out, N_ROWS * Y_DIM);
}
